// MASTER_7602092114466
// MI455X (gfx1250) — matrix operand order repaired and checked by reading, not compiled, not run on hardware
//
#include <hip/hip_runtime.h>
#include <math.h>

// ---------------- types & helpers ----------------
typedef __bf16 v16bf __attribute__((ext_vector_type(16)));
typedef float  v8f   __attribute__((ext_vector_type(8)));
typedef unsigned int u32x4 __attribute__((ext_vector_type(4)));
typedef int i32x4 __attribute__((ext_vector_type(4)));
typedef int i32x8 __attribute__((ext_vector_type(8)));

__device__ __forceinline__ unsigned short f2bf(float f) {
    unsigned int u = __float_as_uint(f);
    unsigned int r = u + 0x7FFFu + ((u >> 16) & 1u);
    return (unsigned short)(r >> 16);
}

// Build a 16-element bf16 fragment from two 16-byte chunks (possibly disjoint).
__device__ __forceinline__ v16bf ld_frag(const unsigned short* p0, const unsigned short* p1) {
    unsigned int t[8];
    *(uint4*)&t[0] = *(const uint4*)p0;
    *(uint4*)&t[4] = *(const uint4*)p1;
    v16bf v;
    __builtin_memcpy(&v, t, sizeof(v));
    return v;
}

__device__ __forceinline__ v8f wmma_bf(v16bf a, v16bf b, v8f c) {
    return __builtin_amdgcn_wmma_f32_16x16x32_bf16(false, a, false, b, (short)0, c, false, false);
}

// LDS byte offset of a __shared__ object (flat shared address, low 32 bits).
__device__ __forceinline__ unsigned lds_off(const void* p) {
    return (unsigned)(unsigned long long)(uintptr_t)p;
}

// ---- Tensor Data Mover: async 2D tile (bf16) global -> LDS ----
// tileW elements per row, tileH rows, row stride strideElems (2-byte elems).
// Writes tile row-major, densely packed (tileH x tileW) at ldsAddr.
__device__ __forceinline__ void tdm_load_2d(unsigned ldsAddr, const void* gptr,
                                            unsigned tileW, unsigned tileH,
                                            unsigned long long strideElems) {
    unsigned long long ga = (unsigned long long)gptr;
    u32x4 g0;
    g0[0] = 1u;                                                   // count=1, user desc
    g0[1] = ldsAddr;                                              // lds_addr (bytes)
    g0[2] = (unsigned)ga;                                         // global_addr[31:0]
    g0[3] = (unsigned)((ga >> 32) & 0x01FFFFFFull) | (2u << 30);  // addr[56:32] | type=2
    unsigned gw[8];
    gw[0] = (1u << 16);                                   // data_size = 1 (2 bytes)
    gw[1] = (tileW & 0xFFFFu) << 16;                      // tensor_dim0[15:0]  @ bit48
    gw[2] = (tileW >> 16) | ((tileH & 0xFFFFu) << 16);    // dim0[31:16] | dim1[15:0]
    gw[3] = (tileH >> 16) | (tileW << 16);                // dim1[31:16] | tile_dim0
    gw[4] = tileH & 0xFFFFu;                              // tile_dim1 (tile_dim2 = 0)
    gw[5] = (unsigned)strideElems;                        // tensor_dim0_stride[31:0]
    gw[6] = (unsigned)(strideElems >> 32) & 0xFFFFu;      // stride[47:32] (dim1_stride=0)
    gw[7] = 0u;
    i32x8 g1;
    __builtin_memcpy(&g1, gw, 32);
    i32x4 z4 = (i32x4)0;
#if __clang_major__ >= 23
    i32x8 z8 = (i32x8)0;
    __builtin_amdgcn_tensor_load_to_lds(g0, g1, z4, z4, z8, 0);
#else
    __builtin_amdgcn_tensor_load_to_lds(g0, g1, z4, z4, 0);
#endif
}

__device__ __forceinline__ void tdm_wait0() {
    __builtin_amdgcn_s_wait_tensorcnt(0);
}

// ---------------- small elementwise / conversion kernels ----------------
__global__ void f32_to_bf16_k(const float* __restrict__ s, unsigned short* __restrict__ d, int n) {
    int i = blockIdx.x * blockDim.x + threadIdx.x;
    if (i < n) d[i] = f2bf(s[i]);
}

// x2y_W (256x158) -> bf16 padded to 256x160
__global__ void pack_x2y_k(const float* __restrict__ w, unsigned short* __restrict__ d) {
    int i = blockIdx.x * blockDim.x + threadIdx.x;
    if (i < 256 * 160) {
        int r = i / 160, c = i % 160;
        d[i] = (c < 158) ? f2bf(w[r * 158 + c]) : (unsigned short)0;
    }
}

// A0[row,k] = bf16(x[row, k] * gate[n, k]), K padded 158->160
__global__ void pack_src_k(const float* __restrict__ x, const float* __restrict__ gate,
                           unsigned short* __restrict__ A0) {
    int i = blockIdx.x * blockDim.x + threadIdx.x;
    if (i < 65536 * 160) {
        int row = i / 160, c = i % 160;
        int n = row >> 6;
        A0[i] = (c < 158) ? f2bf(x[(size_t)row * 221 + c] * gate[(size_t)n * 158 + c])
                          : (unsigned short)0;
    }
}

// temporal: vT[((n*4+h)*64+i)*64 + m] = v[(n*64+m)*256 + h*64 + i]
__global__ void transpose_t_k(const unsigned short* __restrict__ v, unsigned short* __restrict__ vT) {
    int idx = blockIdx.x * blockDim.x + threadIdx.x; // 16777216
    int m = idx & 63, i = (idx >> 6) & 63, h = (idx >> 12) & 3, n = idx >> 14;
    vT[idx] = v[((size_t)((n << 6) + m)) * 256 + (h << 6) + i];
}

// spatial: vT[((t*2+h)*128+d)*1024 + n] = v[(n*64+t)*256 + h*128 + d]
__global__ void transpose_s_k(const unsigned short* __restrict__ v, unsigned short* __restrict__ vT) {
    int idx = blockIdx.x * blockDim.x + threadIdx.x; // 16777216
    int n = idx & 1023, d = (idx >> 10) & 127, h = (idx >> 17) & 1, t = idx >> 18;
    vT[idx] = v[((size_t)((n << 6) + t)) * 256 + (h << 7) + d];
}

// ---------------- gate: softmax(gate_in @ gW^T + gb) * 158 ----------------
__global__ __launch_bounds__(256) void gate_kernel(const float* __restrict__ x,
                                                   const float* __restrict__ gW,
                                                   const float* __restrict__ gb,
                                                   float* __restrict__ gate) {
    __shared__ float gin[64];
    __shared__ float red[256];
    int n = blockIdx.x, j = threadIdx.x;
    if (j < 63) gin[j] = x[((size_t)n * 64 + 63) * 221 + 158 + j];
    __syncthreads();
    float logit = -1e30f;
    if (j < 158) {
        float a = gb[j];
        for (int i = 0; i < 63; i++) a += gin[i] * gW[j * 63 + i];
        logit = a;
    }
    red[j] = logit; __syncthreads();
    for (int s = 128; s > 0; s >>= 1) { if (j < s) red[j] = fmaxf(red[j], red[j + s]); __syncthreads(); }
    float mx = red[0]; __syncthreads();
    float e = (j < 158) ? __expf(logit - mx) : 0.f;
    red[j] = e; __syncthreads();
    for (int s = 128; s > 0; s >>= 1) { if (j < s) red[j] += red[j + s]; __syncthreads(); }
    float inv = 158.f / red[0];
    if (j < 158) gate[(size_t)n * 158 + j] = e * inv;
}

// ---------------- LayerNorm over D=256 (optionally in-place, optional add) ----
__global__ __launch_bounds__(256) void ln_kernel(const float* X, const float* Add,
                                                 const float* __restrict__ g, const float* __restrict__ b,
                                                 float* oF, unsigned short* oB) {
    __shared__ float red[256];
    size_t row = blockIdx.x;
    int d = threadIdx.x;
    float v = X[row * 256 + d];
    if (Add) v += Add[row * 256 + d];
    red[d] = v; __syncthreads();
    for (int s = 128; s > 0; s >>= 1) { if (d < s) red[d] += red[d + s]; __syncthreads(); }
    float mean = red[0] * (1.f / 256.f);
    __syncthreads();
    float c = v - mean;
    red[d] = c * c; __syncthreads();
    for (int s = 128; s > 0; s >>= 1) { if (d < s) red[d] += red[d + s]; __syncthreads(); }
    float var = red[0] * (1.f / 256.f);
    float o = c * rsqrtf(var + 1e-5f) * g[d] + b[d];
    oF[row * 256 + d] = o;
    if (oB) oB[row * 256 + d] = f2bf(o);
}

// ---------------- TDM-fed, double-buffered bf16 WMMA GEMM ----------------
// C = A(MxK) @ W(NcxK)^T + epilogue. Tile 64(M) x 128(N), 256 threads = 8 waves.
// Wave 0 drives the Tensor Data Mover; tiles land densely packed in LDS.
__global__ __launch_bounds__(256) void gemm_bf16(
    const unsigned short* __restrict__ A, int lda,
    const unsigned short* __restrict__ W, int ldw,
    const float* __restrict__ bias, const float* __restrict__ pe,
    const float* resid, float* outF, unsigned short* outB,
    int K, int Nc, int relu) {
    __shared__ __align__(16) unsigned short As[2][64 * 32];
    __shared__ __align__(16) unsigned short Bs[2][128 * 32];
    const int tid = threadIdx.x;
    const int m0 = blockIdx.x * 64;
    const int n0 = blockIdx.y * 128;
    const int wave = tid >> 5, lane = tid & 31;
    const int wm = wave & 3, wn = wave >> 2;
    const int l15 = lane & 15, hi = lane >> 4;
    const int nK = K >> 5;
    v8f acc[4];
    #pragma unroll
    for (int f = 0; f < 4; f++) acc[f] = (v8f){};

    if (wave == 0) {
        tdm_load_2d(lds_off(&As[0][0]), &A[(size_t)m0 * lda], 32, 64, (unsigned long long)lda);
        tdm_load_2d(lds_off(&Bs[0][0]), &W[(size_t)n0 * ldw], 32, 128, (unsigned long long)ldw);
    }
    for (int i = 0; i < nK; i++) {
        if (wave == 0) tdm_wait0();
        __syncthreads(); // tile i ready; all reads of buffer (i&1) from iter i-2 retired
        if (wave == 0 && i + 1 < nK) {
            int k1 = (i + 1) * 32;
            tdm_load_2d(lds_off(&As[(i + 1) & 1][0]), &A[(size_t)m0 * lda + k1], 32, 64, (unsigned long long)lda);
            tdm_load_2d(lds_off(&Bs[(i + 1) & 1][0]), &W[(size_t)n0 * ldw + k1], 32, 128, (unsigned long long)ldw);
        }
        const unsigned short* as = &As[i & 1][0];
        const unsigned short* bs = &Bs[i & 1][0];
        int mo = (wm * 16 + l15) * 32;
        v16bf a = ld_frag(&as[mo + hi * 8], &as[mo + 16 + hi * 8]);
        #pragma unroll
        for (int f = 0; f < 4; f++) {
            int nL = wn * 64 + f * 16 + l15;
            v16bf b = ld_frag(&bs[nL * 32 + hi * 8], &bs[nL * 32 + 16 + hi * 8]);
            acc[f] = wmma_bf(a, b, acc[f]);
        }
    }
    #pragma unroll
    for (int f = 0; f < 4; f++) {
        int n = n0 + wn * 64 + f * 16 + l15;
        #pragma unroll
        for (int r = 0; r < 8; r++) {
            int m = m0 + wm * 16 + hi * 8 + r;
            float v = acc[f][r];
            if (bias) v += bias[n];
            if (pe) v += pe[(m & 63) * 256 + n];
            if (relu) v = v > 0.f ? v : 0.f;
            size_t o = (size_t)m * Nc + n;
            if (resid) v += resid[o];
            if (outF) outF[o] = v;
            if (outB) outB[o] = f2bf(v);
        }
    }
}

// ---------------- generalized two-pass WMMA attention (TDM-staged Q/K) -------
// Token row index in [65536 x 256] activations: row = g*rsG + m*rsM, cols h*DH..
// Temporal: L=64, DH=64, NH=4, rsG=64, rsM=1.  Spatial: L=1024, DH=128, NH=2, rsG=1, rsM=64.
__device__ __forceinline__ void attn_score_tile(
    const unsigned short* qs, const unsigned short* ks, float* sS,
    int DH, int kSteps, int wm, int wn, int l15, int hi, float scale) {
    v8f s0 = (v8f){}, s1 = (v8f){};
    for (int kk = 0; kk < kSteps; kk++) {
        int mo = (wm * 16 + l15) * DH + kk * 32;
        v16bf a = ld_frag(&qs[mo + hi * 8], &qs[mo + 16 + hi * 8]);
        const unsigned short* p0 = &ks[(wn * 32 + l15) * DH + kk * 32 + hi * 8];
        s0 = wmma_bf(a, ld_frag(p0, p0 + 16), s0);
        const unsigned short* p1 = &ks[(wn * 32 + 16 + l15) * DH + kk * 32 + hi * 8];
        s1 = wmma_bf(a, ld_frag(p1, p1 + 16), s1);
    }
    #pragma unroll
    for (int r = 0; r < 8; r++) {
        int m = wm * 16 + hi * 8 + r;
        sS[m * 64 + wn * 32 + l15]      = s0[r] * scale;
        sS[m * 64 + wn * 32 + 16 + l15] = s1[r] * scale;
    }
}

__global__ __launch_bounds__(256) void attn_flash(
    const unsigned short* __restrict__ Q, const unsigned short* __restrict__ Kb,
    const unsigned short* __restrict__ vT, float* __restrict__ O,
    int L, int DH, int NH, int rsG, int rsM, float scale) {
    __shared__ __align__(16) unsigned short qs[64 * 128];
    __shared__ __align__(16) unsigned short ks[64 * 128];
    __shared__ __align__(16) float sS[64 * 64];
    __shared__ __align__(16) unsigned short pb[64 * 64];
    __shared__ float rmax[64];
    __shared__ float rsum[64];
    const int tid = threadIdx.x, lane = tid & 31, wave = tid >> 5;
    const int wm = wave & 3, wn = wave >> 2, l15 = lane & 15, hi = lane >> 4;
    const int g = blockIdx.z, h = blockIdx.y, q0 = blockIdx.x * 64;
    const int colOff = h * DH;
    const int kSteps = DH >> 5;   // 2 (dh=64) or 4 (dh=128)
    const int nkb = L >> 6;       // key blocks of 64
    const size_t rowBaseG = (size_t)g * rsG;
    const unsigned long long rowStride = (unsigned long long)rsM * 256ull;

    if (wave == 0) { // DMA Q tile and first K block into LDS
        tdm_load_2d(lds_off(qs), &Q[(rowBaseG + (size_t)q0 * rsM) * 256 + colOff], DH, 64, rowStride);
        tdm_load_2d(lds_off(ks), &Kb[rowBaseG * 256 + colOff], DH, 64, rowStride);
    }
    if (tid < 64) { rmax[tid] = -1e30f; rsum[tid] = 0.f; }

    // pass 1: row maxima
    for (int kb = 0; kb < nkb; kb++) {
        if (wave == 0) tdm_wait0();
        __syncthreads();
        attn_score_tile(qs, ks, sS, DH, kSteps, wm, wn, l15, hi, scale);
        __syncthreads(); // ks reads done, sS complete
        if (wave == 0 && kb + 1 < nkb)
            tdm_load_2d(lds_off(ks), &Kb[(rowBaseG + (size_t)(kb + 1) * 64 * rsM) * 256 + colOff], DH, 64, rowStride);
        if (tid < 64) {
            float mx = rmax[tid];
            for (int j = 0; j < 64; j++) { float t = sS[tid * 64 + j]; mx = t > mx ? t : mx; }
            rmax[tid] = mx;
        }
    }

    // pass 2: exp, row sums, P @ V
    if (wave == 0 && nkb > 1) // refetch K block 0 (single-block case still holds it)
        tdm_load_2d(lds_off(ks), &Kb[rowBaseG * 256 + colOff], DH, 64, rowStride);
    v8f o[4];
    #pragma unroll
    for (int f = 0; f < 4; f++) o[f] = (v8f){};
    for (int kb = 0; kb < nkb; kb++) {
        if (wave == 0) tdm_wait0();
        __syncthreads();
        attn_score_tile(qs, ks, sS, DH, kSteps, wm, wn, l15, hi, scale);
        __syncthreads(); // ks reads done, sS complete
        if (wave == 0 && kb + 1 < nkb)
            tdm_load_2d(lds_off(ks), &Kb[(rowBaseG + (size_t)(kb + 1) * 64 * rsM) * 256 + colOff], DH, 64, rowStride);
        if (tid < 64) {
            float mx = rmax[tid], s = 0.f;
            for (int j = 0; j < 64; j++) {
                float p = __expf(sS[tid * 64 + j] - mx);
                s += p;
                pb[tid * 64 + j] = f2bf(p);
            }
            rsum[tid] += s;
        }
        __syncthreads(); // pb (and rsum) ready
        for (int kk2 = 0; kk2 < 2; kk2++) {
            int mo = (wm * 16 + l15) * 64 + kk2 * 32;
            v16bf a = ld_frag(&pb[mo + hi * 8], &pb[mo + 16 + hi * 8]);
            for (int f = 0; f < kSteps; f++) {
                int d = wn * (DH >> 1) + f * 16 + l15;
                const unsigned short* p =
                    &vT[(((size_t)g * NH + h) * DH + d) * L + kb * 64 + kk2 * 32 + hi * 8];
                o[f] = wmma_bf(a, ld_frag(p, p + 16), o[f]);
            }
        }
    }
    for (int f = 0; f < kSteps; f++) {
        int d = wn * (DH >> 1) + f * 16 + l15;
        #pragma unroll
        for (int r = 0; r < 8; r++) {
            int m = wm * 16 + hi * 8 + r;
            size_t row = rowBaseG + (size_t)(q0 + m) * rsM;
            O[row * 256 + colOff + d] = o[f][r] / rsum[m];
        }
    }
}

// ---------------- decode: lam = softmax_t(h_t . h_last), out = (sum lam_t y_t) . dW + db
__global__ __launch_bounds__(256) void decode_kernel(const float* __restrict__ h,
                                                     const float* __restrict__ y,
                                                     const float* __restrict__ dW,
                                                     const float* __restrict__ db,
                                                     float* __restrict__ out) {
    __shared__ float hl[256];
    __shared__ float part[256];
    __shared__ float lam[64];
    int n = blockIdx.x, tid = threadIdx.x;
    hl[tid] = h[((size_t)n * 64 + 63) * 256 + tid];
    __syncthreads();
    int tt = tid & 63, c = tid >> 6;
    float p = 0.f;
    const float* hr = &h[((size_t)n * 64 + tt) * 256 + c * 64];
    const float* hc = &hl[c * 64];
    for (int j = 0; j < 64; j++) p += hr[j] * hc[j];
    part[tid] = p;
    __syncthreads();
    if (tid < 64) lam[tid] = part[tid] + part[tid + 64] + part[tid + 128] + part[tid + 192];
    __syncthreads();
    if (tid == 0) {
        float mx = -1e30f;
        for (int t = 0; t < 64; t++) mx = lam[t] > mx ? lam[t] : mx;
        float s = 0.f;
        for (int t = 0; t < 64; t++) { float e = __expf(lam[t] - mx); lam[t] = e; s += e; }
        float inv = 1.f / s;
        for (int t = 0; t < 64; t++) lam[t] *= inv;
    }
    __syncthreads();
    float zd = 0.f;
    for (int t = 0; t < 64; t++) zd += lam[t] * y[((size_t)n * 64 + t) * 256 + tid];
    part[tid] = zd * dW[tid];
    __syncthreads();
    for (int s = 128; s > 0; s >>= 1) { if (tid < s) part[tid] += part[tid + s]; __syncthreads(); }
    if (tid == 0) out[n] = part[0] + db[0];
}

// ---------------- launch ----------------
extern "C" void kernel_launch(void* const* d_in, const int* in_sizes, int n_in,
                              void* d_out, int out_size, void* d_ws, size_t ws_size,
                              hipStream_t stream) {
    (void)in_sizes; (void)n_in; (void)out_size; (void)ws_size;
    const float* x      = (const float*)d_in[0];
    const float* gate_W = (const float*)d_in[1];
    const float* gate_b = (const float*)d_in[2];
    const float* x2y_W  = (const float*)d_in[3];
    const float* x2y_b  = (const float*)d_in[4];
    const float* pe     = (const float*)d_in[5];
    const float* t_qW = (const float*)d_in[6],  *t_kW = (const float*)d_in[7],  *t_vW = (const float*)d_in[8];
    const float* t_n1_g = (const float*)d_in[9],  *t_n1_b = (const float*)d_in[10];
    const float* t_n2_g = (const float*)d_in[11], *t_n2_b = (const float*)d_in[12];
    const float* t_W1 = (const float*)d_in[13], *t_b1 = (const float*)d_in[14];
    const float* t_W2 = (const float*)d_in[15], *t_b2 = (const float*)d_in[16];
    const float* s_qW = (const float*)d_in[17], *s_kW = (const float*)d_in[18], *s_vW = (const float*)d_in[19];
    const float* s_n1_g = (const float*)d_in[20], *s_n1_b = (const float*)d_in[21];
    const float* s_n2_g = (const float*)d_in[22], *s_n2_b = (const float*)d_in[23];
    const float* s_W1 = (const float*)d_in[24], *s_b1 = (const float*)d_in[25];
    const float* s_W2 = (const float*)d_in[26], *s_b2 = (const float*)d_in[27];
    const float* temp_W = (const float*)d_in[28];
    const float* dec_W  = (const float*)d_in[29];
    const float* dec_b  = (const float*)d_in[30];
    float* out = (float*)d_out;

    const size_t NT = 65536; // 1024*64
    char* p = (char*)d_ws;
    auto take = [&](size_t bytes) -> void* {
        void* r = (void*)p;
        p += (bytes + 255) & ~(size_t)255;
        return r;
    };
    float*          y    = (float*)take(NT * 256 * 4);          // residual stream (f32)
    unsigned short* yb   = (unsigned short*)take(NT * 256 * 2); // bf16 shadow
    float*          att  = (float*)take(NT * 256 * 4);          // attn out; later h
    unsigned short* qb   = (unsigned short*)take(NT * 256 * 2); // Q; also A0 & FFN hidden
    unsigned short* kb   = (unsigned short*)take(NT * 256 * 2);
    unsigned short* vb   = (unsigned short*)take(NT * 256 * 2);
    unsigned short* vT   = (unsigned short*)take(NT * 256 * 2);
    float*          gate = (float*)take(1024 * 158 * 4);
    unsigned short* wx2y = (unsigned short*)take(256 * 160 * 2);
    unsigned short* wmat[11];
    for (int i = 0; i < 11; i++) wmat[i] = (unsigned short*)take(256 * 256 * 2);
    unsigned short *wtq = wmat[0], *wtk = wmat[1], *wtv = wmat[2], *wtw1 = wmat[3], *wtw2 = wmat[4];
    unsigned short *wsq = wmat[5], *wsk = wmat[6], *wsv = wmat[7], *wsw1 = wmat[8], *wsw2 = wmat[9];
    unsigned short *wtemp = wmat[10];

    // --- weight conversion (f32 -> bf16) ---
    pack_x2y_k<<<160, 256, 0, stream>>>(x2y_W, wx2y);
    const float* wsrc[11] = {t_qW, t_kW, t_vW, t_W1, t_W2, s_qW, s_kW, s_vW, s_W1, s_W2, temp_W};
    for (int i = 0; i < 11; i++)
        f32_to_bf16_k<<<256, 256, 0, stream>>>(wsrc[i], wmat[i], 65536);

    // --- gate + embed ---
    gate_kernel<<<1024, 256, 0, stream>>>(x, gate_W, gate_b, gate);
    pack_src_k<<<40960, 256, 0, stream>>>(x, gate, qb /*A0*/);
    gemm_bf16<<<dim3(1024, 2), 256, 0, stream>>>(qb, 160, wx2y, 160, x2y_b, pe,
                                                 nullptr, y, nullptr, 160, 256, 0);

    // --- block 1: temporal (heads=4, dh=64, scale=1) ---
    ln_kernel<<<65536, 256, 0, stream>>>(y, nullptr, t_n1_g, t_n1_b, y, yb);
    gemm_bf16<<<dim3(1024, 2), 256, 0, stream>>>(yb, 256, wtq, 256, nullptr, nullptr, nullptr, nullptr, qb, 256, 256, 0);
    gemm_bf16<<<dim3(1024, 2), 256, 0, stream>>>(yb, 256, wtk, 256, nullptr, nullptr, nullptr, nullptr, kb, 256, 256, 0);
    gemm_bf16<<<dim3(1024, 2), 256, 0, stream>>>(yb, 256, wtv, 256, nullptr, nullptr, nullptr, nullptr, vb, 256, 256, 0);
    transpose_t_k<<<65536, 256, 0, stream>>>(vb, vT);
    attn_flash<<<dim3(1, 4, 1024), 256, 0, stream>>>(qb, kb, vT, att, 64, 64, 4, 64, 1, 1.0f);
    ln_kernel<<<65536, 256, 0, stream>>>(y, att, t_n2_g, t_n2_b, y, yb);
    gemm_bf16<<<dim3(1024, 2), 256, 0, stream>>>(yb, 256, wtw1, 256, t_b1, nullptr, nullptr, nullptr, qb, 256, 256, 1);
    gemm_bf16<<<dim3(1024, 2), 256, 0, stream>>>(qb, 256, wtw2, 256, t_b2, nullptr, y, y, nullptr, 256, 256, 0);

    // --- block 2: spatial (heads=2, dh=128, scale=1/sqrt(128)) ---
    ln_kernel<<<65536, 256, 0, stream>>>(y, nullptr, s_n1_g, s_n1_b, y, yb);
    gemm_bf16<<<dim3(1024, 2), 256, 0, stream>>>(yb, 256, wsq, 256, nullptr, nullptr, nullptr, nullptr, qb, 256, 256, 0);
    gemm_bf16<<<dim3(1024, 2), 256, 0, stream>>>(yb, 256, wsk, 256, nullptr, nullptr, nullptr, nullptr, kb, 256, 256, 0);
    gemm_bf16<<<dim3(1024, 2), 256, 0, stream>>>(yb, 256, wsv, 256, nullptr, nullptr, nullptr, nullptr, vb, 256, 256, 0);
    transpose_s_k<<<65536, 256, 0, stream>>>(vb, vT);
    attn_flash<<<dim3(16, 2, 64), 256, 0, stream>>>(qb, kb, vT, att, 1024, 128, 2, 1, 64, 0.08838834764831845f);
    ln_kernel<<<65536, 256, 0, stream>>>(y, att, s_n2_g, s_n2_b, y, yb);
    gemm_bf16<<<dim3(1024, 2), 256, 0, stream>>>(yb, 256, wsw1, 256, s_b1, nullptr, nullptr, nullptr, qb, 256, 256, 1);
    gemm_bf16<<<dim3(1024, 2), 256, 0, stream>>>(qb, 256, wsw2, 256, s_b2, nullptr, y, y, yb, 256, 256, 0);

    // --- decode ---
    gemm_bf16<<<dim3(1024, 2), 256, 0, stream>>>(yb, 256, wtemp, 256, nullptr, nullptr, nullptr, att /*h*/, nullptr, 256, 256, 0);
    decode_kernel<<<1024, 256, 0, stream>>>(att, y, dec_W, dec_b, out);
}
